// CausalSelfAttention_53412213293286
// MI455X (gfx1250) — matrix operand order repaired and checked by reading, not compiled, not run on hardware
//
#include <hip/hip_runtime.h>
#include <hip/hip_bf16.h>
#include <stdint.h>

#define D_MODEL 1024
#define N_HEADS 16
#define HEAD_DIM 64
#define SEQ     2048
#define BATCH   4
#define NROWS   (BATCH * SEQ)   // 8192

typedef __attribute__((ext_vector_type(16))) __bf16 v16bf;
typedef __attribute__((ext_vector_type(8)))  __bf16 v8bf;
typedef __attribute__((ext_vector_type(8)))  float  v8f;
typedef __attribute__((ext_vector_type(4)))  unsigned int v4u;
typedef __attribute__((ext_vector_type(8)))  int v8i;
typedef __attribute__((ext_vector_type(4)))  int v4i;

typedef unsigned short u16;
typedef unsigned int   u32;

// round-to-nearest-even f32 -> bf16 (bit trick)
__device__ __forceinline__ u16 f2bf(float f) {
  u32 u = __float_as_uint(f);
  u32 r = u + 0x7FFFu + ((u >> 16) & 1u);
  return (u16)(r >> 16);
}

__device__ __forceinline__ v16bf combine8x2(uint4 lo, uint4 hi) {
  union { uint4 u; v8bf b; } a, c;
  a.u = lo; c.u = hi;
  v16bf r;
#pragma unroll
  for (int i = 0; i < 8; ++i) { r[i] = a.b[i]; r[i + 8] = c.b[i]; }
  return r;
}

// A-matrix fragment (16x32 bf16), lane = (m = lane&15, h = lane>>4):
//   elements 0..7  <- K = h*8 + (0..7);  elements 8..15 <- K = 16 + h*8 + (0..7)
__device__ __forceinline__ v16bf load_frag_a(const u16* base, int h) {
  uint4 lo = *(const uint4*)(base + h * 8);
  uint4 hi = *(const uint4*)(base + 16 + h * 8);
  return combine8x2(lo, hi);
}

// B-matrix fragment (32x16 bf16), lane = (n = lane&15, h = lane>>4):
//   elements 0..7  <- K = h*8 + (0..7);  elements 8..15 <- K = 16 + h*8 + (0..7)
__device__ __forceinline__ v16bf load_frag_b(const u16* base, int h) {
  uint4 lo = *(const uint4*)(base + h * 8);
  uint4 hi = *(const uint4*)(base + 16 + h * 8);
  return combine8x2(lo, hi);
}

__device__ __forceinline__ v8f wmma_bf16(v16bf a, v16bf b, v8f c) {
  return __builtin_amdgcn_wmma_f32_16x16x32_bf16(false, a, false, b, (short)0, c,
                                                 false, false);
}

// ------------------------------------------------------------------- TDM
// Build a 2-D bf16 tile descriptor (D#) and issue TENSOR_LOAD_TO_LDS.
//   group0: count=1 | lds_addr | global_addr[56:0] | type=2
//   group1: data_size=1(2B) | tensor_dim0/1 | tile_dim0/1 | dim0 stride
// groups 2/3 zero (tile_dim2/3 = 0 -> unused => 2-D transfer).
__device__ __forceinline__ void tdm_load_tile_bf16(
    u32 lds_byte_addr, const void* global_tile,
    u32 tile_d0, u32 tile_d1,            // tile size in elements (d0 = row run)
    u32 tensor_d0, u32 tensor_d1,        // tensor extents (OOB guards)
    unsigned long long stride0_elems) {  // elements between rows
  unsigned long long ga = (unsigned long long)global_tile;
  v4u g0;
  g0[0] = 1u;                                            // count=1, user mode
  g0[1] = lds_byte_addr;
  g0[2] = (u32)ga;                                       // global_addr[31:0]
  g0[3] = (u32)((ga >> 32) & 0x01FFFFFFu) | (2u << 30);  // [56:32] | type=2
  v8i g1;
  g1[0] = (int)(1u << 16);                               // wg_mask=0, data_size=2B
  g1[1] = (int)((tensor_d0 & 0xFFFFu) << 16);            // tensor_dim0[15:0]
  g1[2] = (int)((tensor_d0 >> 16) | ((tensor_d1 & 0xFFFFu) << 16));
  g1[3] = (int)((tensor_d1 >> 16) | (tile_d0 << 16));    // tile_dim0
  g1[4] = (int)(tile_d1 & 0xFFFFu);                      // tile_dim1, tile_dim2=0
  g1[5] = (int)(stride0_elems & 0xFFFFFFFFu);            // dim0 stride [31:0]
  g1[6] = (int)((stride0_elems >> 32) & 0xFFFFu);        // stride[47:32], stride1=0
  g1[7] = 0;
  v4i g2 = {0, 0, 0, 0};
  v4i g3 = {0, 0, 0, 0};
#if __clang_major__ >= 23
  v8i g4 = {0, 0, 0, 0, 0, 0, 0, 0};
  __builtin_amdgcn_tensor_load_to_lds(g0, g1, g2, g3, g4, 0);
#else
  __builtin_amdgcn_tensor_load_to_lds(g0, g1, g2, g3, 0);
#endif
}

__device__ __forceinline__ u32 lds_offset(const void* p) {
  return (u32)(unsigned long long)p;   // LDS aperture: addr[31:0] = LDS offset
}

// ---------------------------------------------------------------- converts
__global__ void cvt_bf16_kernel(const float* __restrict__ src,
                                u16* __restrict__ dst, int n) {
  int i = blockIdx.x * 256 + threadIdx.x;
  if (i < n) dst[i] = f2bf(src[i]);
}

// transpose-convert: dst[n][k] = bf16(src[k][n]) for 1024x1024 weights
__global__ void cvt_bf16_t_kernel(const float* __restrict__ src,
                                  u16* __restrict__ dst) {
  __shared__ float tile[32][33];
  const int k0 = blockIdx.x * 32, n0 = blockIdx.y * 32;
  const int x = threadIdx.x;
#pragma unroll
  for (int i = 0; i < 4; ++i) {
    const int yy = threadIdx.y + i * 8;
    tile[yy][x] = src[(size_t)(k0 + yy) * D_MODEL + n0 + x];
  }
  __syncthreads();
#pragma unroll
  for (int i = 0; i < 4; ++i) {
    const int yy = threadIdx.y + i * 8;
    dst[(size_t)(n0 + yy) * D_MODEL + k0 + x] = f2bf(tile[x][yy]);
  }
}

// ---------------------------------------------------------------- GEMM core
// Block: 256 threads (8 waves, 4x2 wave grid). Block tile 128x128, K-step 32.
// Both operands are row-major bf16 with K contiguous:
//   A  [M][1024]  -> 128x32 tile ; Wt [N][1024] -> 128x32 tile.
// Tiles staged by the Tensor Data Mover, double-buffered; wave 0 drives TDM
// and syncs TENSORcnt; barriers publish the buffer to all 8 waves.
__device__ __forceinline__ void gemm_core(const u16* __restrict__ A,
                                          const u16* __restrict__ Wt,
                                          u16 (*As)[128][32],
                                          u16 (*Bs)[128][32],
                                          int m0, int n0, int tid,
                                          v8f acc[2][4]) {
  const int wave = tid >> 5, lane = tid & 31;
  const int lh = lane >> 4, ln = lane & 15;
  const int wm = wave & 3, wn = wave >> 2;
  const int nIter = D_MODEL / 32;

  const u16* Abase = A + (size_t)m0 * D_MODEL;
  const u16* Bbase = Wt + (size_t)n0 * D_MODEL;

  if (wave == 0) {
    tdm_load_tile_bf16(lds_offset(&As[0][0][0]), Abase, 32, 128,
                       D_MODEL, NROWS, D_MODEL);
    tdm_load_tile_bf16(lds_offset(&Bs[0][0][0]), Bbase, 32, 128,
                       D_MODEL, D_MODEL, D_MODEL);
    __builtin_amdgcn_s_wait_tensorcnt((unsigned short)0);
  }
  __syncthreads();

  for (int it = 0; it < nIter; ++it) {
    const int cur = it & 1;
    if (it + 1 < nIter && wave == 0) {   // prefetch next K-tile via TDM
      const int k0n = (it + 1) * 32;
      tdm_load_tile_bf16(lds_offset(&As[cur ^ 1][0][0]), Abase + k0n, 32, 128,
                         D_MODEL, NROWS, D_MODEL);
      tdm_load_tile_bf16(lds_offset(&Bs[cur ^ 1][0][0]), Bbase + k0n, 32, 128,
                         D_MODEL, D_MODEL, D_MODEL);
    }

    v16bf afr[2], bfr[4];
#pragma unroll
    for (int mt = 0; mt < 2; ++mt)
      afr[mt] = load_frag_a(&As[cur][wm * 32 + mt * 16 + ln][0], lh);
#pragma unroll
    for (int nt = 0; nt < 4; ++nt)
      bfr[nt] = load_frag_b(&Bs[cur][wn * 64 + nt * 16 + ln][0], lh);
#pragma unroll
    for (int mt = 0; mt < 2; ++mt)
#pragma unroll
      for (int nt = 0; nt < 4; ++nt)
        acc[mt][nt] = wmma_bf16(afr[mt], bfr[nt], acc[mt][nt]);

    __syncthreads();                     // everyone done reading buf[cur]
    if (wave == 0 && it + 1 < nIter)
      __builtin_amdgcn_s_wait_tensorcnt((unsigned short)0);
    __syncthreads();                     // next buffer published
  }
}

// ---------------------------------------------------------------- QKV GEMM
// grid.z selects Q/K/V. Q scaled by 1/sqrt(64). Q,K -> [B,H,T,D]; V -> [B,H,D,T].
__global__ __launch_bounds__(256)
void qkv_gemm_kernel(const u16* __restrict__ xb,
                     const u16* __restrict__ wqt, const u16* __restrict__ wkt,
                     const u16* __restrict__ wvt,
                     const float* __restrict__ bq, const float* __restrict__ bk,
                     const float* __restrict__ bv,
                     u16* __restrict__ Qb, u16* __restrict__ Kb,
                     u16* __restrict__ Vtb) {
  const int proj = blockIdx.z;
  const u16*   Wt   = proj == 0 ? wqt : proj == 1 ? wkt : wvt;
  const float* bias = proj == 0 ? bq  : proj == 1 ? bk  : bv;

  const int m0 = blockIdx.x * 128, n0 = blockIdx.y * 128;
  const int tid = threadIdx.x;
  const int wave = tid >> 5, lane = tid & 31;
  const int lh = lane >> 4, ln = lane & 15;
  const int wm = wave & 3, wn = wave >> 2;

  __shared__ __align__(16) u16 As[2][128][32];
  __shared__ __align__(16) u16 Bs[2][128][32];

  v8f acc[2][4];
#pragma unroll
  for (int i = 0; i < 2; ++i)
#pragma unroll
    for (int j = 0; j < 4; ++j)
#pragma unroll
      for (int e = 0; e < 8; ++e) acc[i][j][e] = 0.f;

  gemm_core(xb, Wt, As, Bs, m0, n0, tid, acc);

  const float scale = (proj == 0) ? 0.125f : 1.0f;
#pragma unroll
  for (int mt = 0; mt < 2; ++mt) {
#pragma unroll
    for (int nt = 0; nt < 4; ++nt) {
      const int ng = n0 + wn * 64 + nt * 16 + ln;
      const int h = ng >> 6, d = ng & 63;
      const float bias_n = bias[ng];
#pragma unroll
      for (int r = 0; r < 8; ++r) {
        const int mg = m0 + wm * 32 + mt * 16 + lh * 8 + r;
        const int b = mg >> 11, t = mg & (SEQ - 1);
        const u16 val = f2bf((acc[mt][nt][r] + bias_n) * scale);
        if (proj == 2)
          Vtb[(size_t)((b * N_HEADS + h) * HEAD_DIM + d) * SEQ + t] = val;
        else if (proj == 1)
          Kb[(size_t)((b * N_HEADS + h) * SEQ + t) * HEAD_DIM + d] = val;
        else
          Qb[(size_t)((b * N_HEADS + h) * SEQ + t) * HEAD_DIM + d] = val;
      }
    }
  }
}

// ---------------------------------------------------------------- attention
// Block: 128 threads (4 waves). One (b,h) + 64-row query tile per block.
// Streams 32-key blocks with online softmax (flash attention).
__global__ __launch_bounds__(128)
void attn_kernel(const u16* __restrict__ Qb, const u16* __restrict__ Kb,
                 const u16* __restrict__ Vtb, u16* __restrict__ attb) {
  const int qt = blockIdx.x;   // 64-row query tile
  const int bh = blockIdx.y;   // b*16 + h
  const int b = bh >> 4, h = bh & 15;
  const int wave = threadIdx.x >> 5, lane = threadIdx.x & 31;
  const int lh = lane >> 4, ln = lane & 15;
  const int q0 = qt * 64 + wave * 16;   // this wave's 16 query rows

  __shared__ __align__(16) u16 plds[4][16][32];   // per-wave P transpose buffer

  // Q A-fragments (row = q0 + ln), d-chunks of 32
  const u16* qrow = Qb + (size_t)(bh * SEQ + q0 + ln) * HEAD_DIM;
  v16bf qa[2];
#pragma unroll
  for (int dc = 0; dc < 2; ++dc) qa[dc] = load_frag_a(qrow + dc * 32, lh);

  v8f O[4];
  float mst[8], lst[8];
#pragma unroll
  for (int i = 0; i < 4; ++i)
#pragma unroll
    for (int e = 0; e < 8; ++e) O[i][e] = 0.f;
#pragma unroll
  for (int r = 0; r < 8; ++r) { mst[r] = -1e30f; lst[r] = 0.f; }

  const int jmax = (qt * 64 + 63) >> 5;
  for (int j = 0; j <= jmax; ++j) {
    const int kbase = j * 32;

    // S = Q @ K^T for 32 keys (two 16x16 tiles)
    v8f s[2];
#pragma unroll
    for (int ss = 0; ss < 2; ++ss) {
      v8f z;
#pragma unroll
      for (int e = 0; e < 8; ++e) z[e] = 0.f;
      const u16* krow = Kb + (size_t)(bh * SEQ + kbase + ss * 16 + ln) * HEAD_DIM;
      v16bf kb0 = load_frag_b(krow, lh);
      v16bf kb1 = load_frag_b(krow + 32, lh);
      z = wmma_bf16(qa[0], kb0, z);
      z = wmma_bf16(qa[1], kb1, z);
      s[ss] = z;
    }

    // causal mask + online softmax update; P -> LDS (bf16)
#pragma unroll
    for (int r = 0; r < 8; ++r) {
      const int qg = q0 + lh * 8 + r;
      float v0 = s[0][r], v1 = s[1][r];
      if (kbase + ln > qg)      v0 = -1e30f;
      if (kbase + 16 + ln > qg) v1 = -1e30f;
      float mx = fmaxf(v0, v1);
#pragma unroll
      for (int msk = 1; msk < 16; msk <<= 1)
        mx = fmaxf(mx, __shfl_xor(mx, msk, 32));
      const float nm   = fmaxf(mst[r], mx);
      const float corr = __expf(mst[r] - nm);
      mst[r] = nm;
      const float p0 = __expf(v0 - nm), p1 = __expf(v1 - nm);
      float rs = p0 + p1;
#pragma unroll
      for (int msk = 1; msk < 16; msk <<= 1) rs += __shfl_xor(rs, msk, 32);
      lst[r] = lst[r] * corr + rs;
#pragma unroll
      for (int dblk = 0; dblk < 4; ++dblk) O[dblk][r] *= corr;
      plds[wave][lh * 8 + r][ln]      = f2bf(p0);
      plds[wave][lh * 8 + r][16 + ln] = f2bf(p1);
    }
    // wave-private LDS transpose: ensure DS stores landed before reload
    asm volatile("s_wait_dscnt 0" ::: "memory");
    const v16bf pa = load_frag_a(&plds[wave][ln][0], lh);

    // O += P @ V  (V stored transposed [B,H,D,T] -> contiguous B fragments)
#pragma unroll
    for (int dblk = 0; dblk < 4; ++dblk) {
      const u16* vrow =
          Vtb + (size_t)(bh * HEAD_DIM + dblk * 16 + ln) * SEQ + kbase;
      v16bf vb = load_frag_b(vrow, lh);
      O[dblk] = wmma_bf16(pa, vb, O[dblk]);
    }
  }

  // normalize, scatter to [B,T,C] bf16 for the output projection
#pragma unroll
  for (int r = 0; r < 8; ++r) {
    const int qg = q0 + lh * 8 + r;
    const float inv = 1.0f / lst[r];
    const size_t base = ((size_t)(b * SEQ + qg)) * D_MODEL + h * HEAD_DIM;
#pragma unroll
    for (int dblk = 0; dblk < 4; ++dblk)
      attb[base + dblk * 16 + ln] = f2bf(O[dblk][r] * inv);
  }
}

// ---------------------------------------------------------------- out proj
__global__ __launch_bounds__(256)
void out_gemm_kernel(const u16* __restrict__ A, const u16* __restrict__ Wt,
                     const float* __restrict__ bias, float* __restrict__ out) {
  const int m0 = blockIdx.x * 128, n0 = blockIdx.y * 128;
  const int tid = threadIdx.x;
  const int wave = tid >> 5, lane = tid & 31;
  const int lh = lane >> 4, ln = lane & 15;
  const int wm = wave & 3, wn = wave >> 2;

  __shared__ __align__(16) u16 As[2][128][32];
  __shared__ __align__(16) u16 Bs[2][128][32];

  v8f acc[2][4];
#pragma unroll
  for (int i = 0; i < 2; ++i)
#pragma unroll
    for (int j = 0; j < 4; ++j)
#pragma unroll
      for (int e = 0; e < 8; ++e) acc[i][j][e] = 0.f;

  gemm_core(A, Wt, As, Bs, m0, n0, tid, acc);

#pragma unroll
  for (int mt = 0; mt < 2; ++mt) {
#pragma unroll
    for (int nt = 0; nt < 4; ++nt) {
      const int ng = n0 + wn * 64 + nt * 16 + ln;
      const float bias_n = bias[ng];
#pragma unroll
      for (int r = 0; r < 8; ++r) {
        const int mg = m0 + wm * 32 + mt * 16 + lh * 8 + r;
        out[(size_t)mg * D_MODEL + ng] = acc[mt][nt][r] + bias_n;
      }
    }
  }
}

// ---------------------------------------------------------------- launch
extern "C" void kernel_launch(void* const* d_in, const int* in_sizes, int n_in,
                              void* d_out, int out_size, void* d_ws,
                              size_t ws_size, hipStream_t stream) {
  const float* x  = (const float*)d_in[0];
  const float* wq = (const float*)d_in[1];
  const float* bq = (const float*)d_in[2];
  const float* wk = (const float*)d_in[3];
  const float* bk = (const float*)d_in[4];
  const float* wv = (const float*)d_in[5];
  const float* bv = (const float*)d_in[6];
  const float* wo = (const float*)d_in[7];
  const float* bo = (const float*)d_in[8];
  float* out = (float*)d_out;

  char* ws = (char*)d_ws;
  size_t off = 0;
  const size_t nx   = (size_t)NROWS * D_MODEL;     // 8388608
  const size_t nw   = (size_t)D_MODEL * D_MODEL;   // 1048576
  const size_t nqkv = (size_t)BATCH * N_HEADS * SEQ * HEAD_DIM;  // 8388608

  u16* xb   = (u16*)(ws + off); off += nx * 2;
  u16* wqt  = (u16*)(ws + off); off += nw * 2;   // transposed bf16 weights
  u16* wkt  = (u16*)(ws + off); off += nw * 2;
  u16* wvt  = (u16*)(ws + off); off += nw * 2;
  u16* wot  = (u16*)(ws + off); off += nw * 2;
  u16* Qb   = (u16*)(ws + off); off += nqkv * 2;
  u16* Kb   = (u16*)(ws + off); off += nqkv * 2;
  u16* Vtb  = (u16*)(ws + off); off += nqkv * 2;
  u16* attb = (u16*)(ws + off); off += nx * 2;     // ~88 MB total

  cvt_bf16_kernel<<<(int)((nx + 255) / 256), 256, 0, stream>>>(x, xb, (int)nx);
  dim3 tgrid(D_MODEL / 32, D_MODEL / 32), tblk(32, 8);
  cvt_bf16_t_kernel<<<tgrid, tblk, 0, stream>>>(wq, wqt);
  cvt_bf16_t_kernel<<<tgrid, tblk, 0, stream>>>(wk, wkt);
  cvt_bf16_t_kernel<<<tgrid, tblk, 0, stream>>>(wv, wvt);
  cvt_bf16_t_kernel<<<tgrid, tblk, 0, stream>>>(wo, wot);

  qkv_gemm_kernel<<<dim3(NROWS / 128, D_MODEL / 128, 3), 256, 0, stream>>>(
      xb, wqt, wkt, wvt, bq, bk, bv, Qb, Kb, Vtb);

  attn_kernel<<<dim3(SEQ / 64, BATCH * N_HEADS), 128, 0, stream>>>(Qb, Kb, Vtb,
                                                                   attb);

  out_gemm_kernel<<<dim3(NROWS / 128, D_MODEL / 128), 256, 0, stream>>>(
      attb, wot, bo, out);
}
